// MultiheadAttention_68702296867184
// MI455X (gfx1250) — matrix operand order repaired and checked by reading, not compiled, not run on hardware
//
#include <hip/hip_runtime.h>
#include <hip/hip_bf16.h>
#include <math.h>

typedef __bf16 bf16_t;
typedef __attribute__((ext_vector_type(16))) __bf16 v16bf;
typedef __attribute__((ext_vector_type(8)))  __bf16 v8bf;
typedef __attribute__((ext_vector_type(8)))  float  v8f;

#define T_DIM 1024
#define B_DIM 8
#define E_DIM 512
#define H_DIM 8
#define D_DIM 64
#define BH_DIM 64
#define M_ROWS 8192   // T*B token rows

// ---------------------------------------------------------------------------
// helpers
// ---------------------------------------------------------------------------
__device__ __forceinline__ v8f wmma_bf16(v16bf a, v16bf b, v8f c) {
  // D = A(16x32 bf16) * B(32x16 bf16) + C(16x16 f32)
  return __builtin_amdgcn_wmma_f32_16x16x32_bf16(
      /*neg_a=*/false, a, /*neg_b=*/false, b,
      /*c_mod=*/(short)0, c, /*reuse_a=*/false, /*reuse_b=*/false);
}

__device__ __forceinline__ v16bf cat8(v8bf lo, v8bf hi) {
  v16bf r;
#pragma unroll
  for (int i = 0; i < 8; ++i) { r[i] = lo[i]; r[8 + i] = hi[i]; }
  return r;
}

// ---------------------------------------------------------------------------
// Kernel 0: bulk f32 -> bf16 conversion (8 elements / thread)
// ---------------------------------------------------------------------------
__global__ __launch_bounds__(256) void cvt_bf16_kernel(
    const float* __restrict__ src, bf16_t* __restrict__ dst, int n) {
  const int i = (blockIdx.x * 256 + threadIdx.x) * 8;
  if (i >= n) return;
  float4 a = *(const float4*)(src + i);
  float4 b = *(const float4*)(src + i + 4);
  v8bf r;
  r[0] = (__bf16)a.x; r[1] = (__bf16)a.y; r[2] = (__bf16)a.z; r[3] = (__bf16)a.w;
  r[4] = (__bf16)b.x; r[5] = (__bf16)b.y; r[6] = (__bf16)b.z; r[7] = (__bf16)b.w;
  *(v8bf*)(dst + i) = r;
}

// ---------------------------------------------------------------------------
// Kernel 1: projection  dst = (x @ W^T + b) * scale  (all-bf16 operands).
// x: bf16 row-major [8192, 512] (row = t*B + b); W: bf16 [E,E] row-major.
// vtrans==0 -> dst[bh][t][d] (Q,K) ; vtrans==1 -> dst[bh][d][t] (V, for P@V).
// One wave owns a 32(M) x 64(N) tile; block = 4 waves stacked in M.
// ---------------------------------------------------------------------------
__global__ __launch_bounds__(128) void proj_kernel(
    const bf16_t* __restrict__ x, const bf16_t* __restrict__ w,
    const float* __restrict__ bias, float scale, bf16_t* __restrict__ dst,
    int vtrans) {
  const int lane = threadIdx.x & 31;
  const int wv   = threadIdx.x >> 5;
  const int half = lane >> 4;
  const int l16  = lane & 15;
  const int ntile = blockIdx.x * 64;
  const int mtile = blockIdx.y * 128 + wv * 32;

  v8f acc[2][4];
#pragma unroll
  for (int mf = 0; mf < 2; ++mf)
#pragma unroll
    for (int nf = 0; nf < 4; ++nf) acc[mf][nf] = (v8f){};

  for (int kb = 0; kb < E_DIM; kb += 32) {
    v16bf afrag[2];
#pragma unroll
    for (int mf = 0; mf < 2; ++mf) {
      const bf16_t* p = x + (size_t)(mtile + mf * 16 + l16) * E_DIM + kb + half * 16;
      afrag[mf] = cat8(*(const v8bf*)p, *(const v8bf*)(p + 8));
    }
#pragma unroll
    for (int nf = 0; nf < 4; ++nf) {
      const bf16_t* p = w + (size_t)(ntile + nf * 16 + l16) * E_DIM + kb + half * 16;
      v16bf bfrag = *(const v16bf*)p;
      acc[0][nf] = wmma_bf16(afrag[0], bfrag, acc[0][nf]);
      acc[1][nf] = wmma_bf16(afrag[1], bfrag, acc[1][nf]);
    }
  }

#pragma unroll
  for (int mf = 0; mf < 2; ++mf)
#pragma unroll
    for (int nf = 0; nf < 4; ++nf) {
      const int n = ntile + nf * 16 + l16;
      const float bn = bias[n];
      const int h = n >> 6, d = n & 63;
#pragma unroll
      for (int r = 0; r < 8; ++r) {
        const int m  = mtile + mf * 16 + half * 8 + r;
        const int t  = m >> 3;     // row = t*B + b
        const int bb = m & 7;
        const __bf16 val = (__bf16)((acc[mf][nf][r] + bn) * scale);
        if (vtrans)
          dst[((size_t)(bb * H_DIM + h) * D_DIM + d) * T_DIM + t] = val;
        else
          dst[((size_t)(bb * H_DIM + h) * T_DIM + t) * D_DIM + d] = val;
      }
    }
}

// ---------------------------------------------------------------------------
// Kernel 2: attention for one (bh, 32-query-row) tile.
// Raw scores (32 x 1024 f32) in LDS; softmax writes exp() as bf16 into a
// second LDS buffer (pb) so the P@V loop has zero conversions; P@V reads V
// from the transposed global layout [bh][d][t] (L2-resident). 8 waves/block.
// ---------------------------------------------------------------------------
__global__ __launch_bounds__(256) void attn_kernel(
    const bf16_t* __restrict__ q, const bf16_t* __restrict__ k,
    const bf16_t* __restrict__ vT, const float* __restrict__ bias,
    const float* __restrict__ mask, const unsigned char* __restrict__ kpm,
    bf16_t* __restrict__ o) {
  __shared__ float sc[32 * 1024];                     // 128 KB raw scores
  __shared__ __align__(32) bf16_t pb[32 * 1024];      //  64 KB exp() probs
  __shared__ float rinv[32];

  const int tid  = threadIdx.x;
  const int lane = tid & 31;
  const int wv   = tid >> 5;
  const int half = lane >> 4;
  const int l16  = lane & 15;
  const int bh   = blockIdx.y;
  const int bidx = bh >> 3;                        // batch = bh / H
  const int t0   = blockIdx.x * 32;

  // ---- preload Q fragments for this block's 32 rows: [mf][kstep]
  v16bf qf[2][2];
#pragma unroll
  for (int mf = 0; mf < 2; ++mf)
#pragma unroll
    for (int ks = 0; ks < 2; ++ks) {
      const int trow = t0 + mf * 16 + l16;
      const bf16_t* p = q + ((size_t)bh * T_DIM + trow) * D_DIM + ks * 32 + half * 16;
      qf[mf][ks] = cat8(*(const v8bf*)p, *(const v8bf*)(p + 8));
    }

  // ---- pass 1: scores = Q K^T + bias + mask (+ key-pad -> -inf), into LDS
  for (int st = wv; st < 16; st += 8) {
    const int sbase = st * 64;

    // prefetch this wave's next bias tile (the real HBM stream)
    if (st + 8 < 16) {
      const float* pf = bias + ((size_t)bh * T_DIM + t0 + lane) * T_DIM + (st + 8) * 64;
      __builtin_prefetch(pf, 0, 0);
      __builtin_prefetch(pf + 32, 0, 0);
    }

    v8f acc[2][4];
#pragma unroll
    for (int mf = 0; mf < 2; ++mf)
#pragma unroll
      for (int nf = 0; nf < 4; ++nf) acc[mf][nf] = (v8f){};

#pragma unroll
    for (int ks = 0; ks < 2; ++ks) {
      v16bf bfr[4];
#pragma unroll
      for (int nf = 0; nf < 4; ++nf) {
        const int scol = sbase + nf * 16 + l16;
        bfr[nf] = *(const v16bf*)(k + ((size_t)bh * T_DIM + scol) * D_DIM +
                                  ks * 32 + half * 16);
      }
#pragma unroll
      for (int mf = 0; mf < 2; ++mf)
#pragma unroll
        for (int nf = 0; nf < 4; ++nf)
          acc[mf][nf] = wmma_bf16(qf[mf][ks], bfr[nf], acc[mf][nf]);
    }

#pragma unroll
    for (int mf = 0; mf < 2; ++mf)
#pragma unroll
      for (int nf = 0; nf < 4; ++nf) {
        const int s = sbase + nf * 16 + l16;
        const float kpadd = kpm[bidx * T_DIM + s] ? -1.0e30f : 0.0f;
#pragma unroll
        for (int r = 0; r < 8; ++r) {
          const int m = mf * 16 + half * 8 + r;
          const int t = t0 + m;
          sc[m * 1024 + s] = acc[mf][nf][r] +
                             bias[((size_t)bh * T_DIM + t) * T_DIM + s] +
                             mask[(size_t)t * T_DIM + s] + kpadd;
        }
      }
  }
  __syncthreads();

  // ---- softmax: 8 threads/row; exp(w - max) stored as bf16; rinv = 1/sum
  {
    const int row = tid >> 3, seg = tid & 7;
    const float* pr = &sc[row * 1024 + seg * 128];
    bf16_t* pp = &pb[row * 1024 + seg * 128];
    float mx = -3.0e38f;
#pragma unroll 4
    for (int i = 0; i < 128; ++i) mx = fmaxf(mx, pr[i]);
    for (int o2 = 1; o2 < 8; o2 <<= 1) mx = fmaxf(mx, __shfl_xor(mx, o2, 32));
    float sum = 0.f;
    for (int i = 0; i < 128; i += 8) {
      v8bf c;
#pragma unroll
      for (int j = 0; j < 8; ++j) {
        const float e = __expf(pr[i + j] - mx);
        c[j] = (__bf16)e;
        sum += e;
      }
      *(v8bf*)(pp + i) = c;
    }
    for (int o2 = 1; o2 < 8; o2 <<= 1) sum += __shfl_xor(sum, o2, 32);
    if (seg == 0) rinv[row] = 1.0f / sum;
  }
  __syncthreads();

  // ---- pass 2: O = P V ; each wave owns one 16x16 output fragment.
  // A straight from bf16 LDS probs, B straight from vT[bh][d][t].
  const int mf = wv >> 2, nf = wv & 3;
  const int d  = nf * 16 + l16;
  const bf16_t* vrow = vT + ((size_t)bh * D_DIM + d) * T_DIM;
  const bf16_t* prow = &pb[(mf * 16 + l16) * 1024];
  v8f oacc = (v8f){};
#pragma unroll 4
  for (int kb = 0; kb < 1024; kb += 32) {
    const bf16_t* pr = prow + kb + half * 16;
    v16bf a = cat8(*(const v8bf*)pr, *(const v8bf*)(pr + 8));
    v16bf bfr = *(const v16bf*)(vrow + kb + half * 16);
    oacc = wmma_bf16(a, bfr, oacc);
  }

#pragma unroll
  for (int r = 0; r < 8; ++r) {
    const int m = mf * 16 + half * 8 + r;
    o[((size_t)bh * T_DIM + (t0 + m)) * D_DIM + d] = (__bf16)(oacc[r] * rinv[m]);
  }
}

// ---------------------------------------------------------------------------
// Kernel 3: out = attn @ out_w^T + out_b (f32). attn is bf16 in [bh][t][d];
// token row rm = t*B + b, k = h*64 + d. Output row-major [rm][E] == d_out.
// ---------------------------------------------------------------------------
__global__ __launch_bounds__(128) void out_proj_kernel(
    const bf16_t* __restrict__ a, const bf16_t* __restrict__ w,
    const float* __restrict__ bias, float* __restrict__ out) {
  const int lane = threadIdx.x & 31;
  const int wv   = threadIdx.x >> 5;
  const int half = lane >> 4;
  const int l16  = lane & 15;
  const int ntile = blockIdx.x * 64;
  const int mtile = blockIdx.y * 128 + wv * 32;

  v8f acc[2][4];
#pragma unroll
  for (int mf = 0; mf < 2; ++mf)
#pragma unroll
    for (int nf = 0; nf < 4; ++nf) acc[mf][nf] = (v8f){};

  for (int kb = 0; kb < E_DIM; kb += 32) {
    v16bf afrag[2];
#pragma unroll
    for (int mf = 0; mf < 2; ++mf) {
      const int row = mtile + mf * 16 + l16;
      const int t = row >> 3, bb = row & 7;
      const int k0 = kb + half * 16;       // chunks never cross a head boundary
      const int k1 = kb + half * 16 + 8;
      const bf16_t* p0 = a + ((size_t)(bb * H_DIM + (k0 >> 6)) * T_DIM + t) * D_DIM + (k0 & 63);
      const bf16_t* p1 = a + ((size_t)(bb * H_DIM + (k1 >> 6)) * T_DIM + t) * D_DIM + (k1 & 63);
      afrag[mf] = cat8(*(const v8bf*)p0, *(const v8bf*)p1);
    }
#pragma unroll
    for (int nf = 0; nf < 4; ++nf) {
      const bf16_t* p = w + (size_t)(ntile + nf * 16 + l16) * E_DIM + kb + half * 16;
      v16bf bfrag = *(const v16bf*)p;
      acc[0][nf] = wmma_bf16(afrag[0], bfrag, acc[0][nf]);
      acc[1][nf] = wmma_bf16(afrag[1], bfrag, acc[1][nf]);
    }
  }

#pragma unroll
  for (int mf = 0; mf < 2; ++mf)
#pragma unroll
    for (int nf = 0; nf < 4; ++nf) {
      const int n = ntile + nf * 16 + l16;
      const float bn = bias[n];
#pragma unroll
      for (int r = 0; r < 8; ++r) {
        const int m = mtile + mf * 16 + half * 8 + r;
        out[(size_t)m * E_DIM + n] = acc[mf][nf][r] + bn;
      }
    }
}

// ---------------------------------------------------------------------------
extern "C" void kernel_launch(void* const* d_in, const int* in_sizes, int n_in,
                              void* d_out, int out_size, void* d_ws, size_t ws_size,
                              hipStream_t stream) {
  (void)in_sizes; (void)n_in; (void)out_size; (void)ws_size;
  const float* query      = (const float*)d_in[0];
  const float* attn_bias  = (const float*)d_in[1];
  const float* attn_mask  = (const float*)d_in[2];
  const unsigned char* kp = (const unsigned char*)d_in[3];
  const float* q_w = (const float*)d_in[4];
  const float* q_b = (const float*)d_in[5];
  const float* k_w = (const float*)d_in[6];
  const float* k_b = (const float*)d_in[7];
  const float* v_w = (const float*)d_in[8];
  const float* v_b = (const float*)d_in[9];
  const float* o_w = (const float*)d_in[10];
  const float* o_b = (const float*)d_in[11];

  const size_t NACT = (size_t)M_ROWS * E_DIM;   // 4M elements
  const size_t NWGT = (size_t)E_DIM * E_DIM;    // 256K elements
  bf16_t* qbf = (bf16_t*)d_ws;
  bf16_t* kbf = qbf + NACT;
  bf16_t* vbf = kbf + NACT;   // transposed layout [bh][d][t]
  bf16_t* abf = vbf + NACT;
  bf16_t* xbf = abf + NACT;
  bf16_t* wqb = xbf + NACT;
  bf16_t* wkb = wqb + NWGT;
  bf16_t* wvb = wkb + NWGT;
  bf16_t* wob = wvb + NWGT;

  // precision downcasts (tiny; everything below is then pure-bf16 WMMA feed)
  cvt_bf16_kernel<<<(int)(NACT / (256 * 8)), 256, 0, stream>>>(query, xbf, (int)NACT);
  cvt_bf16_kernel<<<(int)(NWGT / (256 * 8)), 256, 0, stream>>>(q_w, wqb, (int)NWGT);
  cvt_bf16_kernel<<<(int)(NWGT / (256 * 8)), 256, 0, stream>>>(k_w, wkb, (int)NWGT);
  cvt_bf16_kernel<<<(int)(NWGT / (256 * 8)), 256, 0, stream>>>(v_w, wvb, (int)NWGT);
  cvt_bf16_kernel<<<(int)(NWGT / (256 * 8)), 256, 0, stream>>>(o_w, wob, (int)NWGT);

  dim3 gemm_grid(E_DIM / 64, M_ROWS / 128);
  proj_kernel<<<gemm_grid, 128, 0, stream>>>(xbf, wqb, q_b, 0.125f, qbf, 0);
  proj_kernel<<<gemm_grid, 128, 0, stream>>>(xbf, wkb, k_b, 1.0f,   kbf, 0);
  proj_kernel<<<gemm_grid, 128, 0, stream>>>(xbf, wvb, v_b, 1.0f,   vbf, 1);

  attn_kernel<<<dim3(T_DIM / 32, BH_DIM), 256, 0, stream>>>(
      qbf, kbf, vbf, attn_bias, attn_mask, kp, abf);

  out_proj_kernel<<<gemm_grid, 128, 0, stream>>>(abf, wob, o_b, (float*)d_out);
}
